// ContrastiveLoss2Tradition_29953101922500
// MI455X (gfx1250) — matrix operand order repaired and checked by reading, not compiled, not run on hardware
//
#include <hip/hip_runtime.h>
#include <hip/hip_bf16.h>

// ---------------------------------------------------------------------------
// Contrastive loss: z = normalize([ei;ej;ek]) (6144 x 2048), sim = z z^T,
// e = exp(sim/0.1); per-row denom = full sum minus diag, num = block sum minus
// diag; loss = mean over rows of -log((num/cnt)/denom).
// GEMM: bf16 hi/lo split (3x v_wmma_f32_16x16x32_bf16) fused with exp + row
// reduction. Global->LDS staging uses the CDNA5 Tensor Data Mover
// (tensor_load_to_lds + s_wait_tensorcnt) with double-buffered LDS; the TDM
// pad feature reproduces the 80B padded row pitch needed for conflict-free
// ds_load_b128 fragment fetches.
// Workspace: zh (25.17MB) | zl (25.17MB) | denom (24KB) | num (24KB)
// ---------------------------------------------------------------------------

typedef __attribute__((ext_vector_type(16))) __bf16 v16bf;
typedef __attribute__((ext_vector_type(8)))  __bf16 v8bf;
typedef __attribute__((ext_vector_type(8)))  float  v8f;
typedef __attribute__((ext_vector_type(4)))  unsigned int u32x4;
typedef __attribute__((ext_vector_type(8)))  int    i32x8;
typedef __attribute__((ext_vector_type(4)))  int    i32x4;

#define SIZE  6144
#define DDIM  2048
#define BIJ   4096
#define LDSW  40                    // padded row pitch in bf16 (64B data + 16B pad)
#define TILEB (128 * LDSW)          // one tile in bf16 elements (10KB)

__device__ __forceinline__ v8f wmma_bf16(v16bf a, v16bf b, v8f c) {
    return __builtin_amdgcn_wmma_f32_16x16x32_bf16(
        /*neg_a=*/false, a, /*neg_b=*/false, b,
        /*c_mod=*/(short)0, c, /*reuse_a=*/false, /*reuse_b=*/false);
}

// Build a 16-element bf16 fragment from two 16B LDS chunks at p and p+off2.
__device__ __forceinline__ v16bf ld_frag(const __bf16* p, int off2) {
    v16bf r;
    *((v8bf*)&r)     = *(const v8bf*)(p);
    *((v8bf*)&r + 1) = *(const v8bf*)(p + off2);
    return r;
}

// TDM: DMA one 128-row x 32-col bf16 tile (row stride 2048 elems) from global
// to LDS, inserting 16B of padding after every 64B row (pitch 80B).
__device__ __forceinline__ void tdm_load_tile(unsigned lds_addr,
                                              unsigned long long gaddr) {
    u32x4 g0;
    g0[0] = 1u;                                        // count=1, user descriptor
    g0[1] = lds_addr;                                  // LDS byte address
    g0[2] = (unsigned)(gaddr & 0xFFFFFFFFu);           // global_addr[31:0]
    g0[3] = (unsigned)((gaddr >> 32) & 0x01FFFFFFu)    // global_addr[56:32]
          | (2u << 30);                                // type = 2 (image)
    i32x8 g1;
    g1[0] = (int)((1u << 16)                           // data_size: 2 bytes
                | (1u << 20)                           // pad_enable
                | (3u << 22)                           // pad_interval: 16 DWORDs
                | (3u << 25));                         // pad_amount: 4 DWORDs
    g1[1] = (int)((unsigned)DDIM << 16);               // tensor_dim0 = 2048 (lo16)
    g1[2] = (int)((unsigned)SIZE << 16);               // dim0 hi=0 | tensor_dim1 lo16
    g1[3] = (int)(32u << 16);                          // dim1 hi=0 | tile_dim0 = 32
    g1[4] = 128;                                       // tile_dim1 = 128, tile_dim2 = 0
    g1[5] = DDIM;                                      // tensor_dim0_stride = 2048
    g1[6] = 0;                                         // stride hi | dim1_stride lo
    g1[7] = 0;
    i32x4 g2 = {0, 0, 0, 0};                           // 2-D tensor: groups 2/3 unused
    i32x4 g3 = {0, 0, 0, 0};
    i32x8 g4 = {0, 0, 0, 0, 0, 0, 0, 0};               // extra group (clang-23 form)
    __builtin_amdgcn_tensor_load_to_lds(g0, g1, g2, g3, g4, 0);
}

// ---------------------------------------------------------------------------
// Kernel 1: row L2-normalize + bf16 hi/lo split; zero accumulators.
// ---------------------------------------------------------------------------
__global__ __launch_bounds__(256) void nrm_kernel(
    const float* __restrict__ ei, const float* __restrict__ ej,
    const float* __restrict__ ek,
    __bf16* __restrict__ zh, __bf16* __restrict__ zl,
    float* __restrict__ denom, float* __restrict__ num)
{
    const int row = blockIdx.x;
    const int tid = threadIdx.x;
    const float* src = (row < 2048) ? (ei + (size_t)row * DDIM)
                     : (row < 4096) ? (ej + (size_t)(row - 2048) * DDIM)
                                    : (ek + (size_t)(row - 4096) * DDIM);
    float v[8];
    float ss = 0.f;
#pragma unroll
    for (int it = 0; it < 8; ++it) {
        v[it] = src[tid + it * 256];
        ss += v[it] * v[it];
    }
#pragma unroll
    for (int m = 16; m >= 1; m >>= 1) ss += __shfl_xor(ss, m, 32);

    __shared__ float red[8];
    const int lane = tid & 31, wid = tid >> 5;
    if (lane == 0) red[wid] = ss;
    __syncthreads();
    if (tid == 0) {
        float t = 0.f;
#pragma unroll
        for (int i = 0; i < 8; ++i) t += red[i];
        red[0] = 1.0f / fmaxf(sqrtf(t), 1e-12f);
        denom[row] = 0.f;
        num[row]   = 0.f;
    }
    __syncthreads();
    const float rn = red[0];
#pragma unroll
    for (int it = 0; it < 8; ++it) {
        const int idx = row * DDIM + tid + it * 256;
        float x  = v[it] * rn;
        __bf16 h = (__bf16)x;
        zh[idx] = h;
        zl[idx] = (__bf16)(x - (float)h);
    }
}

// ---------------------------------------------------------------------------
// Kernel 2: fused tiled GEMM (128x128 per WG) + exp + row reduction.
// 8 wave32s: wave grid 2(m) x 4(n); each wave owns 64x32 = 4x2 WMMA tiles.
// TDM double-buffered staging: waves 0..3 each DMA one matrix (Ah/Al/Bh/Bl).
// ---------------------------------------------------------------------------
__global__ __launch_bounds__(256) void sim_kernel(
    const __bf16* __restrict__ zh, const __bf16* __restrict__ zl,
    float* __restrict__ denom, float* __restrict__ num)
{
    extern __shared__ __bf16 smem[];   // [2 buf][4 mat][128*LDSW] = 80KB

    const int tid      = threadIdx.x;
    const int rowStart = blockIdx.y * 128;
    const int colStart = blockIdx.x * 128;

    const int lane = tid & 31;
    const int ln15 = lane & 15;
    const int hi   = lane >> 4;           // 0/1: half-wave
    const int w    = tid >> 5;
    const int wm   = w >> 2;              // 0..1  (64-row strips)
    const int wn   = w & 3;               // 0..3  (32-col strips)

    // TDM setup: wave w (<4) owns matrix w: 0=Ah, 1=Al, 2=Bh, 3=Bl.
    const unsigned ldsBase = (unsigned)(uintptr_t)&smem[0];
    unsigned long long gbase = 0;
    unsigned myLds0 = 0, myLds1 = 0;
    if (w < 4) {
        const unsigned long long zb =
            (w & 1) ? (unsigned long long)(uintptr_t)zl
                    : (unsigned long long)(uintptr_t)zh;
        const int rs = (w < 2) ? rowStart : colStart;
        gbase  = zb + (unsigned long long)rs * (DDIM * 2);     // tile row 0, k=0
        myLds0 = ldsBase + (unsigned)((0 * 4 + w) * TILEB) * 2;
        myLds1 = ldsBase + (unsigned)((1 * 4 + w) * TILEB) * 2;
    }

    v8f acc[4][2] = {};

    // Prologue: DMA k-step 0 into buffer 0.
    if (w < 4) tdm_load_tile(myLds0, gbase);

    for (int kt = 0; kt < 64; ++kt) {
        const int cur = kt & 1;
        // Issue DMA for next k-step into the other buffer (its previous
        // consumers finished at the end-of-iteration barrier of kt-1).
        if (w < 4) {
            if (kt < 63) {
                tdm_load_tile(cur ? myLds0 : myLds1,
                              gbase + (unsigned long long)(kt + 1) * 64);
                __builtin_amdgcn_s_wait_tensorcnt(1);   // oldest (cur) done
            } else {
                __builtin_amdgcn_s_wait_tensorcnt(0);
            }
        }
        __syncthreads();                                 // cur buffer visible

        const __bf16* sAh = &smem[(cur * 4 + 0) * TILEB];
        const __bf16* sAl = &smem[(cur * 4 + 1) * TILEB];
        const __bf16* sBh = &smem[(cur * 4 + 2) * TILEB];
        const __bf16* sBl = &smem[(cur * 4 + 3) * TILEB];

#pragma unroll
        for (int nt = 0; nt < 2; ++nt) {
            // B (32x16, K x N): lane = N, K halves split by lane>=16 (matches A).
            const int boff = (wn * 32 + nt * 16 + ln15) * LDSW + hi * 8;
            v16bf bh = ld_frag(&sBh[boff], 16);
            v16bf bl = ld_frag(&sBl[boff], 16);
#pragma unroll
            for (int mt = 0; mt < 4; ++mt) {
                // A (16x32, M x K): lane = M, K halves split by lane>=16.
                const int aoff = (wm * 64 + mt * 16 + ln15) * LDSW + hi * 8;
                v16bf ah = ld_frag(&sAh[aoff], 16);
                v16bf al = ld_frag(&sAl[aoff], 16);
                acc[mt][nt] = wmma_bf16(ah, bh, acc[mt][nt]);  // hi*hi
                acc[mt][nt] = wmma_bf16(ah, bl, acc[mt][nt]);  // hi*lo
                acc[mt][nt] = wmma_bf16(al, bh, acc[mt][nt]);  // lo*hi
            }
        }
        __syncthreads();                                 // done reading cur
    }

    // Epilogue: e = exp2(sim * 10/ln2), mask diagonal, reduce over N,
    // atomically accumulate into denom (always) and num (same block).
    const bool in_block = (rowStart < BIJ) == (colStart < BIJ);
    const float SCL = 14.4269504088896341f;     // 10 / ln(2)
#pragma unroll
    for (int mt = 0; mt < 4; ++mt) {
        float rs[8];
#pragma unroll
        for (int j = 0; j < 8; ++j) rs[j] = 0.f;
#pragma unroll
        for (int nt = 0; nt < 2; ++nt) {
            const int c = colStart + wn * 32 + nt * 16 + ln15;   // N = lane&15
#pragma unroll
            for (int j = 0; j < 8; ++j) {
                const int r = rowStart + wm * 64 + mt * 16 + hi * 8 + j;
                float e = exp2f(acc[mt][nt][j] * SCL);
                if (r == c) e = 0.f;                              // exclude diag
                rs[j] += e;
            }
        }
#pragma unroll
        for (int j = 0; j < 8; ++j) {
            rs[j] += __shfl_xor(rs[j], 1, 32);
            rs[j] += __shfl_xor(rs[j], 2, 32);
            rs[j] += __shfl_xor(rs[j], 4, 32);
            rs[j] += __shfl_xor(rs[j], 8, 32);   // 16-lane (N) reduction
        }
        if (ln15 == 0) {                         // lanes 0 and 16
            const int rbase = rowStart + wm * 64 + mt * 16 + hi * 8;
#pragma unroll
            for (int j = 0; j < 8; ++j) {
                atomicAdd(&denom[rbase + j], rs[j]);
                if (in_block) atomicAdd(&num[rbase + j], rs[j]);
            }
        }
    }
}

// ---------------------------------------------------------------------------
// Kernel 3: final scalar loss.
// ---------------------------------------------------------------------------
__global__ __launch_bounds__(256) void loss_kernel(
    const float* __restrict__ denom, const float* __restrict__ num,
    float* __restrict__ out)
{
    const int tid = threadIdx.x;
    float a = 0.f;
    for (int r = tid; r < SIZE; r += 256) {
        const float cnt = (r < BIJ) ? (float)(BIJ - 1) : (float)(SIZE - BIJ - 1);
        a += -logf(num[r] / (cnt * denom[r]));
    }
#pragma unroll
    for (int m = 16; m >= 1; m >>= 1) a += __shfl_xor(a, m, 32);
    __shared__ float red[8];
    const int lane = tid & 31, wid = tid >> 5;
    if (lane == 0) red[wid] = a;
    __syncthreads();
    if (tid == 0) {
        float t = 0.f;
#pragma unroll
        for (int i = 0; i < 8; ++i) t += red[i];
        out[0] = t / (float)SIZE;
    }
}

// ---------------------------------------------------------------------------
extern "C" void kernel_launch(void* const* d_in, const int* in_sizes, int n_in,
                              void* d_out, int out_size, void* d_ws, size_t ws_size,
                              hipStream_t stream) {
    (void)in_sizes; (void)n_in; (void)out_size; (void)ws_size;
    const float* ei = (const float*)d_in[0];
    const float* ej = (const float*)d_in[1];
    const float* ek = (const float*)d_in[2];

    char* ws = (char*)d_ws;
    const size_t ZBYTES = (size_t)SIZE * DDIM * sizeof(__bf16);  // 25,165,824
    __bf16* zh    = (__bf16*)(ws);
    __bf16* zl    = (__bf16*)(ws + ZBYTES);
    float*  denom = (float*)(ws + 2 * ZBYTES);
    float*  num   = (float*)(ws + 2 * ZBYTES + (size_t)SIZE * sizeof(float));

    nrm_kernel<<<SIZE, 256, 0, stream>>>(ei, ej, ek, zh, zl, denom, num);
    dim3 grid(SIZE / 128, SIZE / 128);
    const size_t ldsBytes = 2 * 4 * (size_t)TILEB * sizeof(__bf16);  // 80KB
    sim_kernel<<<grid, 256, ldsBytes, stream>>>(zh, zl, denom, num);
    loss_kernel<<<1, 256, 0, stream>>>(denom, num, (float*)d_out);
}
